// Sample_16655883174792
// MI455X (gfx1250) — compile-verified, matrix operand order checked by reading, not run on hardware
//
#include <hip/hip_runtime.h>
#include <hip/hip_bf16.h>

// ---------------------------------------------------------------------------
// Point-transformer pipeline for MI455X (gfx1250).
// Hot path (~147 GFLOP of GEMMs) on v_wmma_f32_16x16x32_f16 with LDS-staged
// B tiles via async global->LDS DMA (ASYNCcnt), double buffering, and
// 2 M-tiles per wave for B-fragment register reuse.
// ---------------------------------------------------------------------------

typedef __attribute__((ext_vector_type(16))) _Float16 v16h;
typedef __attribute__((ext_vector_type(8)))  _Float16 v8h;
typedef __attribute__((ext_vector_type(8)))  float    v8f;
typedef __attribute__((ext_vector_type(4)))  int      v4i;

#define NPC   16384
#define NQ    1024
#define BSZ   2
#define KNN   20
#define DIM   128
#define NHEAD 8
#define HDIM  16
#define DFF   2048
#define NLAY  3
#define NSEQ  (BSZ * NQ)          // 2048 sequences
#define MALL  (NSEQ * KNN)        // 40960 rows
#define MCHUNK 4096               // FFN M-chunk

#if __has_builtin(__builtin_amdgcn_global_load_async_to_lds_b128) && \
    __has_builtin(__builtin_amdgcn_s_wait_asynccnt)
#define HAVE_ASYNC_LDS 1
#else
#define HAVE_ASYNC_LDS 0
#endif

// ---------------------------------------------------------------------------
// KNN: one block (256 threads) per query. Per-thread sorted top-20 over a
// strided scan of 16384 points, then LDS tree-merge of sorted lists.
// ---------------------------------------------------------------------------
__global__ __launch_bounds__(256)
void knn_kernel(const float* __restrict__ pc, const float* __restrict__ query,
                float* __restrict__ out_dis, int* __restrict__ out_idx)
{
    __shared__ float sd[256 * KNN];
    __shared__ int   si[256 * KNN];
    const int tid = threadIdx.x;
    const int q   = blockIdx.x;          // 0..2047 (b*NQ + qq)
    const int b   = q >> 10;
    const float qx = query[q * 3 + 0];
    const float qy = query[q * 3 + 1];
    const float qz = query[q * 3 + 2];
    const float* p = pc + (size_t)b * NPC * 3;

    float d[KNN]; int id[KNN];
#pragma unroll
    for (int j = 0; j < KNN; ++j) { d[j] = 3.4e38f; id[j] = 0; }

    for (int i = tid; i < NPC; i += 256) {
        float dx = qx - p[i * 3 + 0];
        float dy = qy - p[i * 3 + 1];
        float dz = qz - p[i * 3 + 2];
        float c = dx * dx + dy * dy + dz * dz;
        if (c < d[KNN - 1]) {
            int j = KNN - 1;
            while (j > 0 && d[j - 1] > c) { d[j] = d[j - 1]; id[j] = id[j - 1]; --j; }
            d[j] = c; id[j] = i;
        }
    }
    for (int j = 0; j < KNN; ++j) { sd[tid * KNN + j] = d[j]; si[tid * KNN + j] = id[j]; }

    for (int stride = 128; stride >= 1; stride >>= 1) {
        __syncthreads();
        if (tid < stride) {
            float a0[KNN]; int a1[KNN]; float b0[KNN]; int b1[KNN];
            for (int j = 0; j < KNN; ++j) {
                a0[j] = sd[tid * KNN + j];            a1[j] = si[tid * KNN + j];
                b0[j] = sd[(tid + stride) * KNN + j]; b1[j] = si[(tid + stride) * KNN + j];
            }
            float m0[KNN]; int m1[KNN];
            int ia = 0, ib = 0;
            for (int j = 0; j < KNN; ++j) {
                bool ta = (ib >= KNN) || (ia < KNN && a0[ia] <= b0[ib]);
                if (ta) { m0[j] = a0[ia]; m1[j] = a1[ia]; ++ia; }
                else    { m0[j] = b0[ib]; m1[j] = b1[ib]; ++ib; }
            }
            for (int j = 0; j < KNN; ++j) { sd[tid * KNN + j] = m0[j]; si[tid * KNN + j] = m1[j]; }
        }
    }
    __syncthreads();
    if (tid < KNN) {
        float v = sd[tid];
        out_dis[q * KNN + tid] = sqrtf(v > 0.0f ? v : 0.0f);
        out_idx[q * KNN + tid] = si[tid];
    }
}

// ---------------------------------------------------------------------------
// Pooled build: one wave per (b,q,k) row. Fuses context assembly, the two
// 7->128 matvecs (scale/shift), and the gather-modulate. Writes f32 + f16.
// ---------------------------------------------------------------------------
__global__ __launch_bounds__(256)
void pooled_kernel(const float* __restrict__ x, const float* __restrict__ pc,
                   const float* __restrict__ query,
                   const float* __restrict__ kw, const float* __restrict__ kb,
                   const float* __restrict__ bw, const float* __restrict__ bb,
                   const float* __restrict__ dis, const int* __restrict__ idx,
                   float* __restrict__ h, _Float16* __restrict__ h16)
{
    const int wave = (blockIdx.x * blockDim.x + threadIdx.x) >> 5;
    const int lane = threadIdx.x & 31;
    if (wave >= MALL) return;
    const int kk = wave % KNN;
    const int q  = wave / KNN;            // b*NQ + qq
    const int b  = q >> 10;
    const int pi = idx[q * KNN + kk];

    float c[7];
    c[0] = dis[q * KNN + kk];
    c[1] = pc[((size_t)b * NPC + pi) * 3 + 0];
    c[2] = pc[((size_t)b * NPC + pi) * 3 + 1];
    c[3] = pc[((size_t)b * NPC + pi) * 3 + 2];
    c[4] = query[q * 3 + 0];
    c[5] = query[q * 3 + 1];
    c[6] = query[q * 3 + 2];

    const float* xr = x + ((size_t)b * NPC + pi) * DIM;
#pragma unroll
    for (int t = 0; t < 4; ++t) {
        const int dc = lane + t * 32;
        float sc = kb[dc], sh = bb[dc];
#pragma unroll
        for (int j = 0; j < 7; ++j) { sc += c[j] * kw[dc * 7 + j]; sh += c[j] * bw[dc * 7 + j]; }
        float v = xr[dc] * sc + sh;
        h  [(size_t)wave * DIM + dc] = v;
        h16[(size_t)wave * DIM + dc] = (_Float16)v;
    }
}

// ---------------------------------------------------------------------------
// WMMA GEMM: C[M,N] = A[M,K] * Bw[N,K]^T + bias, optional ReLU.
// Block = 256 threads = 8 waves; block owns a 256(M) x 64(N) tile; wave w owns
// rows [bm*256 + w*16, +16) and [bm*256 + 128 + w*16, +16) — each B fragment
// read from LDS is reused for 2 WMMAs (halves LDS traffic per FLOP).
// B k-slab (64 cols x 32 k, 4 KB) staged in LDS via async global->LDS DMA
// (double-buffered; ds_store fallback), shared by all 8 waves. A fragments
// double-buffered in registers. Main K-loop is branchless (last step peeled).
// Requires M%256==0, N%64==0, K%32==0 (true for every call site).
//
// Fragment packing per CDNA5 ISA 7.12.2 (16-bit A 16x32; B mirrored):
//   lanes 0-15  : K in {0..7} (v0..v3) and {16..23} (v4..v7)
//   lanes 16-31 : K in {8..15}          and {24..31}
// ---------------------------------------------------------------------------
__global__ __launch_bounds__(256)
void gemm_wmma(const _Float16* __restrict__ A, int lda,
               const _Float16* __restrict__ Bw, int ldb,
               const float* __restrict__ bias,
               float* __restrict__ Cf, _Float16* __restrict__ Ch,
               int M, int N, int Kdim, int relu)
{
    __shared__ __align__(16) _Float16 bs[2][64][32];   // [buf][col][k]
    const int tid  = threadIdx.x;
    const int lane = tid & 31;
    const int w    = tid >> 5;                          // 0..7
    const int nbn  = N >> 6;
    const int bm   = blockIdx.x / nbn;
    const int bn   = blockIdx.x % nbn;
    const int tm   = (bm << 8) + (w << 4);              // first M-tile
    const int tn   = bn << 6;

    // per-thread staging slot: one 16-byte chunk of the 64x32 B slab
    const int scol = tid >> 2;                          // 0..63
    const int sk   = (tid & 3) << 3;                    // 0,8,16,24
    const _Float16* bsrc = Bw + (size_t)(tn + scol) * ldb + sk;

    const int kb0  = (lane < 16) ? 0 : 8;
    const int bcol = (lane & 15);
    const _Float16* aptr0 = A + (size_t)(tm + (lane & 15)) * lda + kb0;
    const _Float16* aptr1 = aptr0 + (size_t)128 * lda;   // second M-tile

    v8f acc0[4], acc1[4];
#pragma unroll
    for (int s = 0; s < 4; ++s) {
        float bv = bias ? bias[tn + bcol + s * 16] : 0.0f;
#pragma unroll
        for (int j = 0; j < 8; ++j) { acc0[s][j] = bv; acc1[s][j] = bv; }
    }

    auto loadA = [&](const _Float16* ap, int k) {
        v8h lo = *(const v8h*)(ap + k);
        v8h hi = *(const v8h*)(ap + k + 16);
        v16h a;
#pragma unroll
        for (int j = 0; j < 8; ++j) { a[j] = lo[j]; a[8 + j] = hi[j]; }
        return a;
    };
    auto mmacs = [&](int cur, const v16h& a0, const v16h& a1) {
#pragma unroll
        for (int s = 0; s < 4; ++s) {
            v8h lo = *(const v8h*)&bs[cur][bcol + s * 16][kb0];
            v8h hi = *(const v8h*)&bs[cur][bcol + s * 16][kb0 + 16];
            v16h bfrag;
#pragma unroll
            for (int j = 0; j < 8; ++j) { bfrag[j] = lo[j]; bfrag[8 + j] = hi[j]; }
            acc0[s] = __builtin_amdgcn_wmma_f32_16x16x32_f16(
                false, a0, false, bfrag, (short)0, acc0[s], false, false);
            acc1[s] = __builtin_amdgcn_wmma_f32_16x16x32_f16(
                false, a1, false, bfrag, (short)0, acc1[s], false, false);
        }
    };

#if HAVE_ASYNC_LDS
    auto stage = [&](int k, int buf) {
        __builtin_amdgcn_global_load_async_to_lds_b128(
            (__attribute__((address_space(1))) v4i*)(bsrc + k),
            (__attribute__((address_space(3))) v4i*)&bs[buf][scol][sk],
            0, 0);
    };
    stage(0, 0);
    v16h a0_cur = loadA(aptr0, 0);
    v16h a1_cur = loadA(aptr1, 0);
    int k = 0;
    for (; k < Kdim - 32; k += 32) {                    // branchless steady state
        const int cur = (k >> 5) & 1;
        stage(k + 32, cur ^ 1);                         // overlap next DMA
        __builtin_amdgcn_s_wait_asynccnt(1);            // oldest slab landed
        __syncthreads();                                // publish slab
        v16h a0_nxt = loadA(aptr0, k + 32);             // overlap A loads
        v16h a1_nxt = loadA(aptr1, k + 32);
        mmacs(cur, a0_cur, a1_cur);
        __syncthreads();                                // reads done before reuse
        a0_cur = a0_nxt; a1_cur = a1_nxt;
    }
    {                                                   // peeled last K-step
        const int cur = (k >> 5) & 1;
        __builtin_amdgcn_s_wait_asynccnt(0);
        __syncthreads();
        mmacs(cur, a0_cur, a1_cur);
    }
#else
    // fallback: regs -> ds_store_b128 staging, single buffer
    for (int k = 0; k < Kdim; k += 32) {
        v8h stg = *(const v8h*)(bsrc + k);
        v16h a0_cur = loadA(aptr0, k);
        v16h a1_cur = loadA(aptr1, k);
        __syncthreads();                                // prior reads done
        *(v8h*)&bs[0][scol][sk] = stg;
        __syncthreads();                                // publish slab
        mmacs(0, a0_cur, a1_cur);
    }
#endif

    const int rbase = tm + ((lane < 16) ? 0 : 8);
#pragma unroll
    for (int s = 0; s < 4; ++s) {
        const int col = tn + bcol + s * 16;
#pragma unroll
        for (int j = 0; j < 8; ++j) {
            float v0 = acc0[s][j];
            float v1 = acc1[s][j];
            if (relu) { v0 = v0 > 0.0f ? v0 : 0.0f; v1 = v1 > 0.0f ? v1 : 0.0f; }
            const size_t off0 = (size_t)(rbase + j) * N + col;
            const size_t off1 = (size_t)(rbase + 128 + j) * N + col;
            if (Cf) { Cf[off0] = v0; Cf[off1] = v1; }
            else    { Ch[off0] = (_Float16)v0; Ch[off1] = (_Float16)v1; }
        }
    }
}

// ---------------------------------------------------------------------------
// Attention: one wave per (sequence, head). K/V staged in LDS; 20x20 scores,
// softmax, weighted sum — all per-lane (seq len 20 is too small for WMMA).
// ---------------------------------------------------------------------------
__global__ __launch_bounds__(256)
void attn_kernel(const _Float16* __restrict__ qkv, _Float16* __restrict__ o16)
{
    __shared__ _Float16 skv[8][2][KNN][HDIM];
    const int tid  = threadIdx.x;
    const int w    = tid >> 5;
    const int lane = tid & 31;
    const int gid  = blockIdx.x * 8 + w;   // 0..NSEQ*NHEAD-1
    const int seq  = gid >> 3;
    const int head = gid & 7;
    const size_t base = (size_t)seq * KNN * (3 * DIM);

    for (int t = lane; t < KNN * HDIM; t += 32) {
        const int r = t >> 4, dc = t & 15;
        skv[w][0][r][dc] = qkv[base + (size_t)r * (3 * DIM) + DIM     + head * HDIM + dc];
        skv[w][1][r][dc] = qkv[base + (size_t)r * (3 * DIM) + 2 * DIM + head * HDIM + dc];
    }
    __syncthreads();

    if (lane < KNN) {
        float qv[HDIM];
#pragma unroll
        for (int dd = 0; dd < HDIM; ++dd)
            qv[dd] = (float)qkv[base + (size_t)lane * (3 * DIM) + head * HDIM + dd];

        float s[KNN];
        float mx = -3.4e38f;
        for (int j = 0; j < KNN; ++j) {
            float a = 0.0f;
#pragma unroll
            for (int dd = 0; dd < HDIM; ++dd) a += qv[dd] * (float)skv[w][0][j][dd];
            s[j] = a * 0.25f;                 // 1/sqrt(16)
            mx = s[j] > mx ? s[j] : mx;
        }
        float sum = 0.0f;
        for (int j = 0; j < KNN; ++j) { s[j] = __expf(s[j] - mx); sum += s[j]; }
        const float inv = 1.0f / sum;
        for (int dd = 0; dd < HDIM; ++dd) {
            float a = 0.0f;
            for (int j = 0; j < KNN; ++j) a += s[j] * (float)skv[w][1][j][dd];
            o16[((size_t)seq * KNN + lane) * DIM + head * HDIM + dd] = (_Float16)(a * inv);
        }
    }
}

// ---------------------------------------------------------------------------
// Residual + LayerNorm: one wave per 128-wide row, shfl_xor wave32 reductions.
// Writes f32 master copy and f16 mirror for the next GEMM.
// ---------------------------------------------------------------------------
__global__ __launch_bounds__(256)
void ln_kernel(const float* __restrict__ hin, const float* __restrict__ add,
               const float* __restrict__ g, const float* __restrict__ bta,
               float* __restrict__ hout, _Float16* __restrict__ h16, int rows)
{
    const int wave = (blockIdx.x * blockDim.x + threadIdx.x) >> 5;
    const int lane = threadIdx.x & 31;
    if (wave >= rows) return;
    const size_t ro = (size_t)wave * DIM;

    float v[4];
    float s = 0.0f;
#pragma unroll
    for (int t = 0; t < 4; ++t) { const int c = lane + t * 32; v[t] = hin[ro + c] + add[ro + c]; s += v[t]; }
#pragma unroll
    for (int m = 16; m >= 1; m >>= 1) s += __shfl_xor(s, m, 32);
    const float mu = s * (1.0f / DIM);

    float vs = 0.0f;
#pragma unroll
    for (int t = 0; t < 4; ++t) { const float d0 = v[t] - mu; vs += d0 * d0; }
#pragma unroll
    for (int m = 16; m >= 1; m >>= 1) vs += __shfl_xor(vs, m, 32);
    const float inv = rsqrtf(vs * (1.0f / DIM) + 1e-5f);

#pragma unroll
    for (int t = 0; t < 4; ++t) {
        const int c = lane + t * 32;
        const float y = (v[t] - mu) * inv * g[c] + bta[c];
        hout[ro + c] = y;
        h16[ro + c]  = (_Float16)y;
    }
}

// ---------------------------------------------------------------------------
__global__ __launch_bounds__(256)
void f32_to_f16(const float* __restrict__ in, _Float16* __restrict__ out, int n)
{
    const int i = blockIdx.x * blockDim.x + threadIdx.x;
    if (i < n) out[i] = (_Float16)in[i];
}

// ---------------------------------------------------------------------------
extern "C" void kernel_launch(void* const* d_in, const int* in_sizes, int n_in,
                              void* d_out, int out_size, void* d_ws, size_t ws_size,
                              hipStream_t stream)
{
    const float* x        = (const float*)d_in[0];
    const float* pc       = (const float*)d_in[1];
    const float* query    = (const float*)d_in[2];
    const float* ctx_k_w  = (const float*)d_in[3];
    const float* ctx_k_b  = (const float*)d_in[4];
    const float* ctx_b_w  = (const float*)d_in[5];
    const float* ctx_b_b  = (const float*)d_in[6];
    const float* attn_in_w  = (const float*)d_in[7];
    const float* attn_in_b  = (const float*)d_in[8];
    const float* attn_out_w = (const float*)d_in[9];
    const float* attn_out_b = (const float*)d_in[10];
    const float* ffn_w1   = (const float*)d_in[11];
    const float* ffn_b1   = (const float*)d_in[12];
    const float* ffn_w2   = (const float*)d_in[13];
    const float* ffn_b2   = (const float*)d_in[14];
    const float* ln1_g    = (const float*)d_in[15];
    const float* ln1_b    = (const float*)d_in[16];
    const float* ln2_g    = (const float*)d_in[17];
    const float* ln2_b    = (const float*)d_in[18];
    const float* lin_w    = (const float*)d_in[19];
    const float* lin_b    = (const float*)d_in[20];

    // --- workspace carve-out ---
    char* ws = (char*)d_ws;
    size_t off = 0;
    auto carve = [&](size_t bytes) -> char* {
        off = (off + 255) & ~(size_t)255;
        char* p = ws + off;
        off += bytes;
        return p;
    };
    float*    knn_dis = (float*)   carve((size_t)MALL * 4);
    int*      knn_idx = (int*)     carve((size_t)MALL * 4);
    float*    h       = (float*)   carve((size_t)MALL * DIM * 4);
    _Float16* h16     = (_Float16*)carve((size_t)MALL * DIM * 2);
    _Float16* qkv16   = (_Float16*)carve((size_t)MALL * 3 * DIM * 2);
    _Float16* o16     = (_Float16*)carve((size_t)MALL * DIM * 2);
    float*    tmp32   = (float*)   carve((size_t)MALL * DIM * 4);
    _Float16* ff16    = (_Float16*)carve((size_t)MCHUNK * DFF * 2);
    _Float16* w_ain   = (_Float16*)carve((size_t)NLAY * 3 * DIM * DIM * 2);
    _Float16* w_aout  = (_Float16*)carve((size_t)NLAY * DIM * DIM * 2);
    _Float16* w_f1    = (_Float16*)carve((size_t)NLAY * DFF * DIM * 2);
    _Float16* w_f2    = (_Float16*)carve((size_t)NLAY * DIM * DFF * 2);
    _Float16* w_lin   = (_Float16*)carve((size_t)DIM * KNN * DIM * 2);
    (void)ws_size; (void)in_sizes; (void)n_in; (void)out_size;

    auto conv = [&](const float* src, _Float16* dst, int n) {
        f32_to_f16<<<(n + 255) / 256, 256, 0, stream>>>(src, dst, n);
    };
    auto gemm = [&](const _Float16* A, int lda, const _Float16* Bw, int ldb,
                    const float* bias, float* Cf, _Float16* Ch,
                    int M, int N, int Kd, int relu) {
        const int blocks = (M / 256) * (N / 64);   // exact (M%256==0, N%64==0)
        gemm_wmma<<<blocks, 256, 0, stream>>>(A, lda, Bw, ldb, bias, Cf, Ch, M, N, Kd, relu);
    };

    // 0) weight conversion f32 -> f16 (deterministic; runs every call)
    conv(attn_in_w,  w_ain,  NLAY * 3 * DIM * DIM);
    conv(attn_out_w, w_aout, NLAY * DIM * DIM);
    conv(ffn_w1,     w_f1,   NLAY * DFF * DIM);
    conv(ffn_w2,     w_f2,   NLAY * DIM * DFF);
    conv(lin_w,      w_lin,  DIM * KNN * DIM);

    // 1) KNN + pooled features
    knn_kernel<<<NSEQ, 256, 0, stream>>>(pc, query, knn_dis, knn_idx);
    pooled_kernel<<<(MALL * 32) / 256, 256, 0, stream>>>(
        x, pc, query, ctx_k_w, ctx_k_b, ctx_b_w, ctx_b_b, knn_dis, knn_idx, h, h16);

    // 2) transformer layers
    for (int i = 0; i < NLAY; ++i) {
        // qkv projection: [MALL,128] x [384,128]^T -> f16 [MALL,384]
        gemm(h16, DIM, w_ain + (size_t)i * 3 * DIM * DIM, DIM,
             attn_in_b + (size_t)i * 3 * DIM, nullptr, qkv16, MALL, 3 * DIM, DIM, 0);

        // attention per (seq, head)
        attn_kernel<<<NSEQ, 256, 0, stream>>>(qkv16, o16);

        // output projection -> f32 [MALL,128]
        gemm(o16, DIM, w_aout + (size_t)i * DIM * DIM, DIM,
             attn_out_b + (size_t)i * DIM, tmp32, nullptr, MALL, DIM, DIM, 0);

        // h = LN(h + attn_out)
        ln_kernel<<<(MALL * 32) / 256, 256, 0, stream>>>(
            h, tmp32, ln1_g + (size_t)i * DIM, ln1_b + (size_t)i * DIM, h, h16, MALL);

        // FFN in M-chunks (keeps the f16 intermediate at 16.8 MB)
        for (int c = 0; c < MALL / MCHUNK; ++c) {
            gemm(h16 + (size_t)c * MCHUNK * DIM, DIM,
                 w_f1 + (size_t)i * DFF * DIM, DIM,
                 ffn_b1 + (size_t)i * DFF, nullptr, ff16, MCHUNK, DFF, DIM, 1);
            gemm(ff16, DFF,
                 w_f2 + (size_t)i * DIM * DFF, DFF,
                 ffn_b2 + (size_t)i * DIM, tmp32 + (size_t)c * MCHUNK * DIM, nullptr,
                 MCHUNK, DIM, DFF, 0);
        }

        // h = LN(h + ffn_out)
        ln_kernel<<<(MALL * 32) / 256, 256, 0, stream>>>(
            h, tmp32, ln2_g + (size_t)i * DIM, ln2_b + (size_t)i * DIM, h, h16, MALL);
    }

    // 3) final projection: [2048,2560] x [128,2560]^T -> d_out f32 [2048,128]
    gemm(h16, KNN * DIM, w_lin, KNN * DIM, lin_b, (float*)d_out, nullptr,
         NSEQ, DIM, KNN * DIM, 0);
}
